// CrossAttention_40630390620806
// MI455X (gfx1250) — compile-verified, matrix operand order checked by reading, not run on hardware
//
#include <hip/hip_runtime.h>

#define B_   4
#define H_   8
#define NQ_  2048
#define NKV_ 2048
#define CM_  8
#define CS_  16
#define DPAD 160          // 128 mv + 16 scalar + 16 pad
#define SCALE (1.0f/12.0f) // 1/sqrt(16*CM + CS) = 1/sqrt(144)

typedef __bf16 v16bf __attribute__((ext_vector_type(16)));
typedef __bf16 v8bf  __attribute__((ext_vector_type(8)));
typedef float  v8f   __attribute__((ext_vector_type(8)));

__constant__ int   c_GRADE[16] = {0,1,1,1,1,2,2,2,2,2,2,3,3,3,3,4};
__constant__ float c_ETA[16]   = {1.f,1.f,-1.f,-1.f,-1.f,-1.f,-1.f,-1.f,
                                  1.f,1.f,1.f,1.f,1.f,1.f,-1.f,-1.f};

static __device__ inline v16bf pack16(v8bf lo, v8bf hi) {
  v16bf r;
#pragma unroll
  for (int e = 0; e < 8; ++e) { r[e] = lo[e]; r[e + 8] = hi[e]; }
  return r;
}

// Async DMA: global -> LDS, 16 bytes per lane, tracked by ASYNCcnt (gfx1250).
static __device__ inline void async_ld16(const __bf16* gsrc, __bf16* lds_dst) {
  unsigned int lds_off = (unsigned int)(unsigned long long)lds_dst; // low 32b = LDS byte addr
  asm volatile("global_load_async_to_lds_b128 %0, %1, off"
               :: "v"(lds_off), "v"(gsrc) : "memory");
}
static __device__ inline void wait_async0() {
  asm volatile("s_wait_asynccnt 0x0" ::: "memory");
}

// ---------------- Q projection: equi_linear -> bf16 features [b][h][n][DPAD]
__global__ __launch_bounds__(256)
void proj_q_kernel(const float* __restrict__ mv, const float* __restrict__ s,
                   const float* __restrict__ w_mv, const float* __restrict__ w_s2mv,
                   const float* __restrict__ w_mv2s, const float* __restrict__ w_s,
                   __bf16* __restrict__ qf)
{
  const int tok = blockIdx.x;           // b*NQ + n
  const int b = tok / NQ_, n = tok % NQ_;
  __shared__ float smv[256];
  __shared__ float ss[32];
  smv[threadIdx.x] = mv[(size_t)tok * 256 + threadIdx.x];
  if (threadIdx.x < 32) ss[threadIdx.x] = s[(size_t)tok * 32 + threadIdx.x];
  __syncthreads();
#pragma unroll
  for (int u = 0; u < 5; ++u) {
    int idx = threadIdx.x * 5 + u;      // 0..1279  (H * DPAD)
    int h = idx / DPAD;
    int d = idx % DPAD;
    float val = 0.f;
    if (d < 128) {
      int c = d >> 4, comp = d & 15;
      int o = c * H_ + h;
      const float* w = w_mv + c_GRADE[comp] * (64 * 16) + o * 16;
#pragma unroll
      for (int i = 0; i < 16; ++i) val += smv[i * 16 + comp] * w[i];
      if (comp == 0) {
        const float* w2 = w_s2mv + o * 32;
#pragma unroll
        for (int j = 0; j < 32; ++j) val += ss[j] * w2[j];
      }
      val *= SCALE * c_ETA[comp];
    } else if (d < 144) {
      int c = d - 128, o = c * H_ + h;
      const float* w1 = w_s + o * 32;
#pragma unroll
      for (int j = 0; j < 32; ++j) val += ss[j] * w1[j];
      const float* w2 = w_mv2s + o * 16;
#pragma unroll
      for (int i = 0; i < 16; ++i) val += smv[i * 16] * w2[i];
      val *= SCALE;
    }
    qf[(((size_t)b * H_ + h) * NQ_ + n) * DPAD + d] = (__bf16)val;
  }
}

// ---------------- KV projection -> kf [b][h][n][DPAD], vf transposed [b][h][d][n]
//                  head_scale folded into vf
__global__ __launch_bounds__(256)
void proj_kv_kernel(const float* __restrict__ mv, const float* __restrict__ s,
                    const float* __restrict__ w_mv, const float* __restrict__ w_s2mv,
                    const float* __restrict__ w_mv2s, const float* __restrict__ w_s,
                    const float* __restrict__ hscale,
                    __bf16* __restrict__ kf, __bf16* __restrict__ vf)
{
  const int tok = blockIdx.x;
  const int b = tok / NKV_, n = tok % NKV_;
  __shared__ float smv[256];
  __shared__ float ss[32];
  smv[threadIdx.x] = mv[(size_t)tok * 256 + threadIdx.x];
  if (threadIdx.x < 32) ss[threadIdx.x] = s[(size_t)tok * 32 + threadIdx.x];
  __syncthreads();
#pragma unroll
  for (int u = 0; u < 10; ++u) {
    int idx = threadIdx.x * 10 + u;     // 0..2559
    int kv  = idx / (H_ * DPAD);        // 0=k, 1=v
    int rem = idx % (H_ * DPAD);
    int h = rem / DPAD;
    int d = rem % DPAD;
    float val = 0.f;
    if (d < 128) {
      int c = d >> 4, comp = d & 15;
      int o = kv * 64 + c * H_ + h;     // channel in 2*H*CM = 128
      const float* w = w_mv + c_GRADE[comp] * (128 * 16) + o * 16;
#pragma unroll
      for (int i = 0; i < 16; ++i) val += smv[i * 16 + comp] * w[i];
      if (comp == 0) {
        const float* w2 = w_s2mv + o * 32;
#pragma unroll
        for (int j = 0; j < 32; ++j) val += ss[j] * w2[j];
      }
    } else if (d < 144) {
      int c = d - 128;
      int o = kv * 128 + c * H_ + h;    // channel in 2*H*CS = 256
      const float* w1 = w_s + o * 32;
#pragma unroll
      for (int j = 0; j < 32; ++j) val += ss[j] * w1[j];
      const float* w2 = w_mv2s + o * 16;
#pragma unroll
      for (int i = 0; i < 16; ++i) val += smv[i * 16] * w2[i];
    }
    if (kv == 0) {
      kf[(((size_t)b * H_ + h) * NKV_ + n) * DPAD + d] = (__bf16)val;
    } else {
      // transposed: vf[b][h][d][n]
      vf[(((size_t)b * H_ + h) * DPAD + d) * NKV_ + n] = (__bf16)(val * hscale[h]);
    }
  }
}

// ---------------- Flash attention: one wave = 16 queries, block = 8 waves,
//                  K/V tiles staged in LDS via async DMA
__global__ __launch_bounds__(256)
void attn_kernel(const __bf16* __restrict__ qf, const __bf16* __restrict__ kf,
                 const __bf16* __restrict__ vf, float* __restrict__ hout)
{
  const int qtile = blockIdx.x;     // 0..15 (128 queries)
  const int h = blockIdx.y, b = blockIdx.z;
  const int wave = threadIdx.x >> 5;
  const int lane = threadIdx.x & 31;
  const int lh = lane >> 4;         // half (0/1)
  const int ln = lane & 15;

  const size_t bh = (size_t)b * H_ + h;
  const __bf16* qbase = qf + (bh * NQ_ + (size_t)qtile * 128 + wave * 16) * DPAD;
  const __bf16* kbase = kf + bh * NKV_ * DPAD;          // [key][d]
  const __bf16* vbase = vf + bh * DPAD * NKV_;          // [d][key]

  __shared__ __bf16 Kt[32 * DPAD];      // row-major: 32 keys x DPAD
  __shared__ __bf16 Vt[DPAD * 32];      // Vt[d][key]
  __shared__ __bf16 Pst[8][16 * 32];    // per-wave P restage

  // Q A-fragments (16 x 32 bf16 each), 5 chunks covering DPAD=160
  v16bf aq[5];
  {
    const __bf16* qrow = qbase + (size_t)ln * DPAD;
#pragma unroll
    for (int ck = 0; ck < 5; ++ck) {
      v8bf lo = *(const v8bf*)(qrow + ck * 32 + 8 * lh);
      v8bf hi = *(const v8bf*)(qrow + ck * 32 + 16 + 8 * lh);
      aq[ck] = pack16(lo, hi);
    }
  }

  v8f o_acc[10];
  float mrow[8], lrow[8];
#pragma unroll
  for (int dg = 0; dg < 10; ++dg)
#pragma unroll
    for (int c = 0; c < 8; ++c) o_acc[dg][c] = 0.f;
#pragma unroll
  for (int c = 0; c < 8; ++c) { mrow[c] = -3.0e38f; lrow[c] = 0.f; }

  for (int j = 0; j < NKV_; j += 32) {
    __syncthreads();   // previous tile fully consumed
    // K tile: 32*DPAD bf16 = 640 x 16B chunks, contiguous
    for (int c = threadIdx.x; c < (32 * DPAD) / 8; c += 256)
      async_ld16(kbase + (size_t)j * DPAD + c * 8, Kt + c * 8);
    // V tile: DPAD rows x 32 keys, each row 4 x 16B chunks from d-major vf
    for (int c = threadIdx.x; c < (DPAD * 32) / 8; c += 256) {
      int d = c >> 2, part = c & 3;
      async_ld16(vbase + (size_t)d * NKV_ + j + part * 8, Vt + d * 32 + part * 8);
    }
    wait_async0();
    __syncthreads();   // DMA'd data visible to all waves

    // scores: two 16-key halves, 5 K-chunks each
    v8f sfr[2];
#pragma unroll
    for (int nh = 0; nh < 2; ++nh) {
      const __bf16* krow = Kt + (nh * 16 + ln) * DPAD;
      v8f acc = {0.f, 0.f, 0.f, 0.f, 0.f, 0.f, 0.f, 0.f};
#pragma unroll
      for (int ck = 0; ck < 5; ++ck) {
        v8bf lo = *(const v8bf*)(krow + ck * 32 + 8 * lh);
        v8bf hi = *(const v8bf*)(krow + ck * 32 + 16 + 8 * lh);
        v16bf bfrag = pack16(lo, hi);
        acc = __builtin_amdgcn_wmma_f32_16x16x32_bf16(
            false, aq[ck], false, bfrag, (short)0, acc, false, false);
      }
      sfr[nh] = acc;
    }

    // online softmax per row (row = c + 8*lh; cols across 16 lanes x 2 halves)
#pragma unroll
    for (int c = 0; c < 8; ++c) {
      float rm = fmaxf(sfr[0][c], sfr[1][c]);
      rm = fmaxf(rm, __shfl_xor(rm, 1));
      rm = fmaxf(rm, __shfl_xor(rm, 2));
      rm = fmaxf(rm, __shfl_xor(rm, 4));
      rm = fmaxf(rm, __shfl_xor(rm, 8));
      float mnew = fmaxf(mrow[c], rm);
      float corr = __expf(mrow[c] - mnew);
      mrow[c] = mnew;
      float p0 = __expf(sfr[0][c] - mnew);
      float p1 = __expf(sfr[1][c] - mnew);
      float rs = p0 + p1;
      rs += __shfl_xor(rs, 1);
      rs += __shfl_xor(rs, 2);
      rs += __shfl_xor(rs, 4);
      rs += __shfl_xor(rs, 8);
      lrow[c] = lrow[c] * corr + rs;
#pragma unroll
      for (int dg = 0; dg < 10; ++dg) o_acc[dg][c] *= corr;
      // restage P into A-fragment layout (per-wave LDS; DS ops in-order per wave)
      Pst[wave][(c + 8 * lh) * 32 + ln]      = (__bf16)p0;
      Pst[wave][(c + 8 * lh) * 32 + 16 + ln] = (__bf16)p1;
    }

    // P as A-fragment (16 x 32)
    v16bf ap;
    {
      const __bf16* prow = &Pst[wave][ln * 32];
      v8bf lo = *(const v8bf*)(prow + 8 * lh);
      v8bf hi = *(const v8bf*)(prow + 16 + 8 * lh);
      ap = pack16(lo, hi);
    }

    // O += P @ V : 10 d-groups of 16 cols
#pragma unroll
    for (int dg = 0; dg < 10; ++dg) {
      const __bf16* vcol = Vt + (dg * 16 + ln) * 32;
      v8bf lo = *(const v8bf*)(vcol + 8 * lh);
      v8bf hi = *(const v8bf*)(vcol + 16 + 8 * lh);
      v16bf bfrag = pack16(lo, hi);
      o_acc[dg] = __builtin_amdgcn_wmma_f32_16x16x32_bf16(
          false, ap, false, bfrag, (short)0, o_acc[dg], false, false);
    }
  }

  // finalize: divide by row sums, write fp32 [b][h][q][DPAD]
  float* obase = hout + (bh * NQ_ + (size_t)qtile * 128 + wave * 16) * DPAD;
#pragma unroll
  for (int c = 0; c < 8; ++c) {
    float inv = 1.0f / lrow[c];
    int row = c + 8 * lh;
#pragma unroll
    for (int dg = 0; dg < 10; ++dg)
      obase[(size_t)row * DPAD + dg * 16 + ln] = o_acc[dg][c] * inv;
  }
}

// ---------------- output projection: merge heads + equi_linear -> (mv 16x16, s 32)
__global__ __launch_bounds__(256)
void out_proj_kernel(const float* __restrict__ hout,
                     const float* __restrict__ w_mv, const float* __restrict__ w_s2mv,
                     const float* __restrict__ w_mv2s, const float* __restrict__ w_s,
                     float* __restrict__ out_mv, float* __restrict__ out_s)
{
  const int tok = blockIdx.x;
  const int b = tok / NQ_, n = tok % NQ_;
  __shared__ float hmv[64 * 16];
  __shared__ float hs[128];
  for (int t = threadIdx.x; t < 64 * 16; t += 256) {
    int i = t >> 4, x = t & 15;
    int h = i >> 3, c = i & 7;                  // merged channel = h*CM + c
    hmv[t] = hout[(((size_t)b * H_ + h) * NQ_ + n) * DPAD + c * 16 + x];
  }
  if (threadIdx.x < 128) {
    int j = threadIdx.x;
    int h = j >> 4, c = j & 15;                 // merged scalar = h*CS + c
    hs[j] = hout[(((size_t)b * H_ + h) * NQ_ + n) * DPAD + 128 + c];
  }
  __syncthreads();

  {
    int o = threadIdx.x >> 4, x = threadIdx.x & 15;
    const float* w = w_mv + c_GRADE[x] * (16 * 64) + o * 64;
    float val = 0.f;
#pragma unroll
    for (int i = 0; i < 64; ++i) val += hmv[i * 16 + x] * w[i];
    if (x == 0) {
      const float* w2 = w_s2mv + o * 128;
#pragma unroll
      for (int j = 0; j < 128; ++j) val += hs[j] * w2[j];
    }
    out_mv[((size_t)tok * 16 + o) * 16 + x] = val;
  }
  if (threadIdx.x < 32) {
    int o = threadIdx.x;
    const float* w1 = w_s + o * 128;
    const float* w2 = w_mv2s + o * 64;
    float val = 0.f;
#pragma unroll
    for (int j = 0; j < 128; ++j) val += hs[j] * w1[j];
#pragma unroll
    for (int i = 0; i < 64; ++i) val += hmv[i * 16] * w2[i];
    out_s[(size_t)tok * 32 + o] = val;
  }
}

extern "C" void kernel_launch(void* const* d_in, const int* in_sizes, int n_in,
                              void* d_out, int out_size, void* d_ws, size_t ws_size,
                              hipStream_t stream) {
  const float* mv_kv    = (const float*)d_in[0];
  const float* mv_q     = (const float*)d_in[1];
  const float* s_kv     = (const float*)d_in[2];
  const float* s_q      = (const float*)d_in[3];
  const float* w_mv_q   = (const float*)d_in[4];
  const float* w_s2mv_q = (const float*)d_in[5];
  const float* w_mv2s_q = (const float*)d_in[6];
  const float* w_s_q    = (const float*)d_in[7];
  const float* w_mv_kv  = (const float*)d_in[8];
  const float* w_s2mv_kv= (const float*)d_in[9];
  const float* w_mv2s_kv= (const float*)d_in[10];
  const float* w_s_kv   = (const float*)d_in[11];
  const float* w_mv_o   = (const float*)d_in[12];
  const float* w_s2mv_o = (const float*)d_in[13];
  const float* w_mv2s_o = (const float*)d_in[14];
  const float* w_s_o    = (const float*)d_in[15];
  const float* hscale   = (const float*)d_in[16];

  const size_t featElems = (size_t)B_ * H_ * NQ_ * DPAD;   // 10.5M
  char* ws = (char*)d_ws;
  __bf16* qf = (__bf16*)ws;
  __bf16* kf = (__bf16*)(ws + featElems * 2);
  __bf16* vf = (__bf16*)(ws + featElems * 4);
  float*  ho = (float*)(ws + featElems * 6);

  float* out_mv = (float*)d_out;
  float* out_s  = out_mv + (size_t)B_ * NQ_ * 16 * 16;

  proj_q_kernel<<<B_ * NQ_, 256, 0, stream>>>(mv_q, s_q, w_mv_q, w_s2mv_q,
                                              w_mv2s_q, w_s_q, qf);
  proj_kv_kernel<<<B_ * NKV_, 256, 0, stream>>>(mv_kv, s_kv, w_mv_kv, w_s2mv_kv,
                                                w_mv2s_kv, w_s_kv, hscale, kf, vf);
  attn_kernel<<<dim3(NQ_ / 128, H_, B_), 256, 0, stream>>>(qf, kf, vf, ho);
  out_proj_kernel<<<B_ * NQ_, 256, 0, stream>>>(ho, w_mv_o, w_s2mv_o,
                                                w_mv2s_o, w_s_o, out_mv, out_s);
}
